// MyModel_87522843559297
// MI455X (gfx1250) — matrix operand order repaired and checked by reading, not compiled, not run on hardware
//
#include <hip/hip_runtime.h>
#include <hip/hip_bf16.h>

// ---------------------------------------------------------------------------
// Seq2seq beam-search inference for MI455X (gfx1250, wave32, WMMA).
// All dense GEMMs run on v_wmma_f32_16x16x32_bf16 with f32 accumulation.
// K-chunks are double-buffered in LDS: A tiles via global_load_async_to_lds_b128
// (ASYNCcnt), B tiles via register-staged transpose, overlapped with WMMA.
// ---------------------------------------------------------------------------

typedef __bf16 bf16_t;
typedef __attribute__((ext_vector_type(16))) __bf16 v16bf;
typedef __attribute__((ext_vector_type(8)))  __bf16 bf16x8;
typedef __attribute__((ext_vector_type(2)))  __bf16 bf16x2;
typedef __attribute__((ext_vector_type(8)))  float  v8f;

#define NB   32      // batch
#define NS   128     // encoder seq len
#define NE   512     // embedding dim
#define NH   512     // hidden dim
#define NA   512     // attention dim
#define NV   32000   // vocab
#define NK   4       // beam width
#define NT   48      // output length
#define NM   (NB*NK) // 128 decoder rows
#define EOSID 2
#define STARTID 1
#define NEGINF (-1e9f)

static __device__ __forceinline__ bf16_t tobf(float f) { return (bf16_t)f; }

// ---------------------------------------------------------------------------
// Generic bf16 WMMA GEMM: C[M,N](f32) = A[M,K](bf16,lda) * B[K,N](bf16,ldb)
//                                        (+ Cin[M,N](f32,ldcin)) (+ bias[N])
// REQUIRES: M % 16 == 0, N % 64 == 0, K % 64 == 0 (true for every call here).
// Block = 128 threads = 4 waves; block tile 16(M) x 64(N); K chunk = 64.
// Double-buffered LDS: chunk c+1 staged (async A DMA + B reg loads) while the
// two WMMAs of chunk c execute; one barrier per chunk.
// ---------------------------------------------------------------------------
__global__ __launch_bounds__(128)
void gemm_bf16_wmma(const bf16_t* __restrict__ A, int lda,
                    const bf16_t* __restrict__ Bm, int ldb,
                    float* __restrict__ C, int ldc,
                    const float* __restrict__ Cin, int ldcin,
                    const float* __restrict__ bias,
                    int M, int N, int K)
{
    __shared__ bf16_t sA[2][16][72];   // 16 x 64 used (+pad, rows 144B)
    __shared__ bf16_t sB[2][64][72];   // transposed: sB[.][n][k]

    const int tid  = threadIdx.x;
    const int wave = tid >> 5;
    const int lane = tid & 31;
    const int m0 = blockIdx.y * 16;
    const int n0 = blockIdx.x * 64;

    // A staging coords: 16 x 64 tile = 128 lanes x 16B async DMA
    const int ar  = tid >> 3;          // 0..15
    const int akk = (tid & 7) * 8;     // 0,8,..,56
    // B staging coords: thread owns column pair np, k rows bk0 + 4*i
    const int bnp = (tid & 31) * 2;    // 0,2,..,62
    const int bk0 = tid >> 5;          // 0..3
    // fragment coords
    const int arow = lane & 15;
    const int ak   = (lane < 16) ? 0 : 8;     // A frag: 16-bit 16x32 layout
    const int bcol = wave * 16 + (lane & 15);
    const int bk   = (lane < 16) ? 0 : 8;     // B frag: same 16-bit 16x32 layout as A

    v8f acc = {};

    // ---- prologue: stage chunk 0 into buffer 0 ----
    {
        const bf16_t* gp = A + (size_t)(m0 + ar) * lda + akk;
        unsigned lp = (unsigned)(size_t)&sA[0][ar][akk];
        asm volatile("global_load_async_to_lds_b128 %0, %1, off"
                     :: "v"(lp), "v"(gp) : "memory");
        const bf16_t* gb = Bm + (size_t)bk0 * ldb + (n0 + bnp);
        bf16x2 vb[16];
#pragma unroll
        for (int i = 0; i < 16; ++i) vb[i] = *(const bf16x2*)(gb + (size_t)4 * i * ldb);
#pragma unroll
        for (int i = 0; i < 16; ++i) {
            sB[0][bnp][bk0 + 4 * i]     = vb[i][0];
            sB[0][bnp + 1][bk0 + 4 * i] = vb[i][1];
        }
        asm volatile("s_wait_asynccnt 0x0" ::: "memory");
        __syncthreads();
    }

    const int nchunk = K >> 6;
    for (int c = 0; c < nchunk; ++c) {
        const int  cur  = c & 1;
        const int  nxt  = cur ^ 1;
        const bool more = (c + 1 < nchunk);

        // ---- stage chunk c+1: async A DMA now, B global loads into regs ----
        bf16x2 vb[16];
        if (more) {
            const int k0 = (c + 1) << 6;
            const bf16_t* gp = A + (size_t)(m0 + ar) * lda + (k0 + akk);
            unsigned lp = (unsigned)(size_t)&sA[nxt][ar][akk];
            asm volatile("global_load_async_to_lds_b128 %0, %1, off"
                         :: "v"(lp), "v"(gp) : "memory");
            const bf16_t* gb = Bm + (size_t)(k0 + bk0) * ldb + (n0 + bnp);
#pragma unroll
            for (int i = 0; i < 16; ++i) vb[i] = *(const bf16x2*)(gb + (size_t)4 * i * ldb);
            if (c + 2 < nchunk)  // L2 prefetch of the chunk after next
                __builtin_prefetch(gb + (size_t)64 * ldb, 0, 1);
        }

        // ---- compute: two K=32 WMMAs from current buffer ----
#pragma unroll
        for (int h = 0; h < 2; ++h) {
            const int kb = h * 32;
            v16bf av, bv;
#pragma unroll
            for (int i = 0; i < 8; ++i) {
                av[i]     = sA[cur][arow][kb + ak + i];
                av[8 + i] = sA[cur][arow][kb + ak + 16 + i];
            }
#pragma unroll
            for (int i = 0; i < 8; ++i) {
                bv[i]     = sB[cur][bcol][kb + bk + i];
                bv[8 + i] = sB[cur][bcol][kb + bk + 16 + i];
            }
            acc = __builtin_amdgcn_wmma_f32_16x16x32_bf16(
                      false, av, false, bv, (short)0, acc, false, false);
        }

        // ---- finish staging chunk c+1: spill B regs to LDS, wait DMA ----
        if (more) {
#pragma unroll
            for (int i = 0; i < 16; ++i) {
                sB[nxt][bnp][bk0 + 4 * i]     = vb[i][0];
                sB[nxt][bnp + 1][bk0 + 4 * i] = vb[i][1];
            }
            asm volatile("s_wait_asynccnt 0x0" ::: "memory");
        }
        __syncthreads();
    }

    // ---- epilogue: f32 C/D 16x16 layout (VGPR r: M = r or 8+r) ----
    const int col   = lane & 15;
    const int rbase = (lane < 16) ? 0 : 8;
    const int n     = n0 + wave * 16 + col;
#pragma unroll
    for (int r = 0; r < 8; ++r) {
        int m = m0 + rbase + r;
        float v = acc[r];
        if (Cin)  v += Cin[(size_t)m * ldcin + n];
        if (bias) v += bias[n];
        C[(size_t)m * ldc + n] = v;
    }
}

// ---------------------------------------------------------------------------
// Utility / pointwise kernels
// ---------------------------------------------------------------------------
__global__ void fill_zero_u32(unsigned int* p, int n) {
    int i = blockIdx.x * 256 + threadIdx.x;
    if (i < n) p[i] = 0u;
}

__global__ void cvt_f32_to_bf16(const float* __restrict__ in, bf16_t* __restrict__ out, int n) {
    int i = blockIdx.x * 256 + threadIdx.x;
    if (i < n) out[i] = tobf(in[i]);
}

__global__ void enc_embed(const int* __restrict__ ids, const float* __restrict__ emb,
                          bf16_t* __restrict__ xe, int n) {
    int i = blockIdx.x * 256 + threadIdx.x;
    if (i >= n) return;
    int e  = i & (NE - 1);
    int rs = i >> 9;                 // NE == 512
    xe[i] = tobf(emb[(size_t)ids[rs] * NE + e]);
}

// LSTM pointwise: gates i,f,g,o in z[M,4H]; c update; h = o*tanh(c)
__global__ void lstm_cell(const float* __restrict__ z, const float* __restrict__ bias,
                          const float* __restrict__ cin, float* __restrict__ cout,
                          float* __restrict__ hout, int ldh,
                          bf16_t* __restrict__ hbf, int ldhb,
                          int M, int H) {
    int i = blockIdx.x * 256 + threadIdx.x;
    if (i >= M * H) return;
    int m  = i / H;
    int hh = i - m * H;
    const float* zr = z + (size_t)m * 4 * H;
    float ig = 1.f / (1.f + __expf(-(zr[hh]          + bias[hh])));
    float fg = 1.f / (1.f + __expf(-(zr[H + hh]      + bias[H + hh])));
    float gg = tanhf(zr[2 * H + hh] + bias[2 * H + hh]);
    float og = 1.f / (1.f + __expf(-(zr[3 * H + hh]  + bias[3 * H + hh])));
    float cn = fg * cin[i] + ig * gg;
    float hn = og * tanhf(cn);
    cout[i] = cn;
    hout[(size_t)m * ldh + hh] = hn;
    if (hbf) hbf[(size_t)m * ldhb + hh] = tobf(hn);
}

// Luong dot scores + softmax fused: one block per (b,k); thread s over S.
__global__ __launch_bounds__(128)
void attn_scores_softmax(const float* __restrict__ cell, const float* __restrict__ keys,
                         float* __restrict__ align) {
    int row = blockIdx.x;            // b*NK + k
    int b   = row >> 2;
    int s   = threadIdx.x;           // 0..127
    const float* cr = cell + (size_t)row * NH;
    const float* kr = keys + ((size_t)b * NS + s) * NA;
    float acc = 0.f;
    for (int h = 0; h < NH; ++h) acc += cr[h] * kr[h];

    __shared__ float red[NS];
    red[s] = acc; __syncthreads();
    for (int off = 64; off; off >>= 1) {
        if (s < off) red[s] = fmaxf(red[s], red[s + off]);
        __syncthreads();
    }
    float mx = red[0]; __syncthreads();
    float ex = __expf(acc - mx);
    red[s] = ex; __syncthreads();
    for (int off = 64; off; off >>= 1) {
        if (s < off) red[s] += red[s + off];
        __syncthreads();
    }
    align[(size_t)row * NS + s] = ex / red[0];
}

// ctx = align @ enc_out, fused with building cat2 = bf16(concat(cell, ctx))
__global__ void ctx_cat(const float* __restrict__ align, const float* __restrict__ encout,
                        const float* __restrict__ cell, bf16_t* __restrict__ cat2) {
    int i = blockIdx.x * 256 + threadIdx.x;  // NM*NH
    if (i >= NM * NH) return;
    int row = i >> 9;
    int h   = i & (NH - 1);
    int b   = row >> 2;
    const float* al = align + (size_t)row * NS;
    const float* eo = encout + (size_t)b * NS * NH + h;
    float acc = 0.f;
    for (int s = 0; s < NS; ++s) acc += al[s] * eo[(size_t)s * NH];
    cat2[(size_t)row * (2 * NH) + h]      = tobf(cell[i]);
    cat2[(size_t)row * (2 * NH) + NH + h] = tobf(acc);
}

// in-place log-softmax over rows of length NV
__global__ __launch_bounds__(256)
void log_softmax_rows(float* __restrict__ x) {
    int row = blockIdx.x;
    int t   = threadIdx.x;
    float* r = x + (size_t)row * NV;
    __shared__ float red[256];
    float mx = -3e38f;
    for (int i = t; i < NV; i += 256) mx = fmaxf(mx, r[i]);
    red[t] = mx; __syncthreads();
    for (int off = 128; off; off >>= 1) {
        if (t < off) red[t] = fmaxf(red[t], red[t + off]);
        __syncthreads();
    }
    mx = red[0]; __syncthreads();
    float sm = 0.f;
    for (int i = t; i < NV; i += 256) sm += __expf(r[i] - mx);
    red[t] = sm; __syncthreads();
    for (int off = 128; off; off >>= 1) {
        if (t < off) red[t] += red[t + off];
        __syncthreads();
    }
    float lse = mx + __logf(red[0]);
    for (int i = t; i < NV; i += 256) r[i] -= lse;
}

// top-4 over K*V candidates per batch; writes new cum/fin/tok + (parent,tok) trace
__global__ __launch_bounds__(256)
void beam_topk(const float* __restrict__ lp, const float* __restrict__ cum_in,
               const int* __restrict__ fin_in,
               float* __restrict__ cum_out, int* __restrict__ fin_out,
               int* __restrict__ tok_out,
               int* __restrict__ parents_t, int* __restrict__ toks_t) {
    int b = blockIdx.x;
    int t = threadIdx.x;
    float bv[NK]; int bi[NK];
#pragma unroll
    for (int j = 0; j < NK; ++j) { bv[j] = -3e38f; bi[j] = 0; }

    for (int c = t; c < NK * NV; c += 256) {
        int k = c / NV;
        int v = c - k * NV;
        float cl  = cum_in[b * NK + k];
        float lpv = fin_in[b * NK + k] ? ((v == EOSID) ? 0.f : NEGINF)
                                       : lp[((size_t)(b * NK + k)) * NV + v];
        float val = cl + lpv;
        if (val > bv[NK - 1]) {
            int pos = NK - 1;
            while (pos > 0 && val > bv[pos - 1]) {
                bv[pos] = bv[pos - 1]; bi[pos] = bi[pos - 1]; --pos;
            }
            bv[pos] = val; bi[pos] = c;
        }
    }
    __shared__ float sv[256 * NK];
    __shared__ int   si[256 * NK];
#pragma unroll
    for (int j = 0; j < NK; ++j) { sv[t * NK + j] = bv[j]; si[t * NK + j] = bi[j]; }
    __syncthreads();
    if (t == 0) {
        for (int j = 0; j < NK; ++j) {
            float best = -3e38f; int bidx = 0, bpos = 0;
            for (int q = 0; q < 256 * NK; ++q) {
                if (sv[q] > best) { best = sv[q]; bidx = si[q]; bpos = q; }
            }
            sv[bpos] = -3.5e38f;
            int par = bidx / NV;
            int tk  = bidx - par * NV;
            cum_out[b * NK + j]   = best;
            fin_out[b * NK + j]   = fin_in[b * NK + par] | (tk == EOSID);
            tok_out[b * NK + j]   = tk;
            parents_t[b * NK + j] = par;
            toks_t[b * NK + j]    = tk;
        }
    }
}

// reorder beam state by parent index
__global__ void beam_gather(const int* __restrict__ parents_t,
                            const float* __restrict__ cell, const float* __restrict__ cnew,
                            const float* __restrict__ attnnew,
                            float* __restrict__ c_state, float* __restrict__ attn_state,
                            bf16_t* __restrict__ hbf) {
    int i = blockIdx.x * 256 + threadIdx.x;  // NM*NH
    if (i >= NM * NH) return;
    int row = i >> 9;
    int h   = i & (NH - 1);
    int b   = row >> 2, k = row & 3;
    int p   = parents_t[b * NK + k];
    int src = (b * NK + p) * NH + h;
    float hv = cell[src];
    hbf[i]        = tobf(hv);
    c_state[i]    = cnew[src];
    attn_state[i] = attnnew[src];
}

// build decoder cell input: inp = bf16(concat(dec_emb[tok], attn_state))
__global__ void dec_embed_cat(const int* __restrict__ tok, const float* __restrict__ demb,
                              const float* __restrict__ attn_state, bf16_t* __restrict__ inp) {
    int i = blockIdx.x * 256 + threadIdx.x;  // NM*(NE+NA)
    if (i >= NM * (NE + NA)) return;
    int row = i >> 10;
    int e   = i & 1023;
    bf16_t v;
    if (e < NE) v = tobf(demb[(size_t)tok[row] * NE + e]);
    else        v = tobf(attn_state[(size_t)row * NA + (e - NE)]);
    inp[i] = v;
}

// beam-search init: tile encoder final state over beams
__global__ void dec_init(const float* __restrict__ encout, const float* __restrict__ cenc,
                         bf16_t* __restrict__ hbf, float* __restrict__ c_state,
                         float* __restrict__ attn_state,
                         float* __restrict__ cum, int* __restrict__ fin, int* __restrict__ tok) {
    int i = blockIdx.x * 256 + threadIdx.x;
    if (i < NM * NH) {
        int row = i >> 9;
        int h   = i & (NH - 1);
        int b   = row >> 2;
        float hv = encout[((size_t)b * NS + (NS - 1)) * NH + h];
        hbf[i]        = tobf(hv);
        c_state[i]    = cenc[b * NH + h];
        attn_state[i] = 0.f;
    }
    if (i < NM) {
        int k = i & 3;
        cum[i] = (k == 0) ? 0.f : NEGINF;
        fin[i] = 0;
        tok[i] = STARTID;
    }
}

// gather_tree backtrack: out[b][t][k]
__global__ void gather_tree(const int* __restrict__ parents, const int* __restrict__ toks,
                            int* __restrict__ out) {
    int i = threadIdx.x;
    if (i >= NM) return;
    int b = i >> 2, k = i & 3;
    int ptr = k;
    for (int t = NT - 1; t >= 0; --t) {
        out[(b * NT + t) * NK + k] = toks[t * NM + b * NK + ptr];
        ptr = parents[t * NM + b * NK + ptr];
    }
}

// ---------------------------------------------------------------------------
// Host orchestration
// ---------------------------------------------------------------------------
static inline char* wsoff(void* base, size_t& off, size_t bytes) {
    char* p = (char*)base + off;
    off += (bytes + 255) & ~(size_t)255;
    return p;
}

extern "C" void kernel_launch(void* const* d_in, const int* in_sizes, int n_in,
                              void* d_out, int out_size, void* d_ws, size_t ws_size,
                              hipStream_t stream) {
    (void)in_sizes; (void)n_in; (void)out_size; (void)ws_size;

    const int*   enc_ids = (const int*)  d_in[0];
    // d_in[1] decoder_inputs: unused by the beam-search reference
    const float* enc_emb = (const float*)d_in[2];
    const float* dec_emb = (const float*)d_in[3];
    const float* W_enc   = (const float*)d_in[4];
    const float* U_enc   = (const float*)d_in[5];
    const float* b_enc   = (const float*)d_in[6];
    const float* W_dec   = (const float*)d_in[7];
    const float* U_dec   = (const float*)d_in[8];
    const float* b_dec   = (const float*)d_in[9];
    const float* W_mem   = (const float*)d_in[10];
    const float* W_attn  = (const float*)d_in[11];
    const float* W_out   = (const float*)d_in[12];
    const float* b_out   = (const float*)d_in[13];

    size_t off = 0;
    bf16_t* Wencb   = (bf16_t*)wsoff(d_ws, off, (size_t)NE * 4 * NH * 2);
    bf16_t* Uencb   = (bf16_t*)wsoff(d_ws, off, (size_t)NH * 4 * NH * 2);
    bf16_t* Wdecb   = (bf16_t*)wsoff(d_ws, off, (size_t)(NE + NA) * 4 * NH * 2);
    bf16_t* Udecb   = (bf16_t*)wsoff(d_ws, off, (size_t)NH * 4 * NH * 2);
    bf16_t* Wmemb   = (bf16_t*)wsoff(d_ws, off, (size_t)NH * NA * 2);
    bf16_t* Wattnb  = (bf16_t*)wsoff(d_ws, off, (size_t)2 * NH * NA * 2);
    bf16_t* Woutb   = (bf16_t*)wsoff(d_ws, off, (size_t)NA * NV * 2);
    bf16_t* xe      = (bf16_t*)wsoff(d_ws, off, (size_t)NB * NS * NE * 2);
    float*  gatesx  = (float*) wsoff(d_ws, off, (size_t)NB * NS * 4 * NH * 4);
    bf16_t* encoutb = (bf16_t*)wsoff(d_ws, off, (size_t)NB * NS * NH * 2);
    float*  encout  = (float*) wsoff(d_ws, off, (size_t)NB * NS * NH * 4);
    float*  keys    = (float*) wsoff(d_ws, off, (size_t)NB * NS * NA * 4);
    bf16_t* zerobf  = (bf16_t*)wsoff(d_ws, off, (size_t)NB * NH * 2);
    float*  z_enc   = (float*) wsoff(d_ws, off, (size_t)NB * 4 * NH * 4);
    float*  c_enc   = (float*) wsoff(d_ws, off, (size_t)NB * NH * 4);
    bf16_t* hbf     = (bf16_t*)wsoff(d_ws, off, (size_t)NM * NH * 2);
    float*  c_state = (float*) wsoff(d_ws, off, (size_t)NM * NH * 4);
    float*  c_new   = (float*) wsoff(d_ws, off, (size_t)NM * NH * 4);
    float*  attn_st = (float*) wsoff(d_ws, off, (size_t)NM * NA * 4);
    float*  cellout = (float*) wsoff(d_ws, off, (size_t)NM * NH * 4);
    float*  attnnew = (float*) wsoff(d_ws, off, (size_t)NM * NA * 4);
    bf16_t* attnnb  = (bf16_t*)wsoff(d_ws, off, (size_t)NM * NA * 2);
    bf16_t* inp     = (bf16_t*)wsoff(d_ws, off, (size_t)NM * (NE + NA) * 2);
    bf16_t* cat2    = (bf16_t*)wsoff(d_ws, off, (size_t)NM * 2 * NH * 2);
    float*  z_dec   = (float*) wsoff(d_ws, off, (size_t)NM * 4 * NH * 4);
    float*  align   = (float*) wsoff(d_ws, off, (size_t)NM * NS * 4);
    float*  logits  = (float*) wsoff(d_ws, off, (size_t)NM * NV * 4);
    float*  cum0    = (float*) wsoff(d_ws, off, (size_t)NM * 4);
    float*  cum1    = (float*) wsoff(d_ws, off, (size_t)NM * 4);
    int*    fin0    = (int*)   wsoff(d_ws, off, (size_t)NM * 4);
    int*    fin1    = (int*)   wsoff(d_ws, off, (size_t)NM * 4);
    int*    tok     = (int*)   wsoff(d_ws, off, (size_t)NM * 4);
    int*    parents = (int*)   wsoff(d_ws, off, (size_t)NT * NM * 4);
    int*    toks    = (int*)   wsoff(d_ws, off, (size_t)NT * NM * 4);

    dim3 blk(256);
    auto g1 = [](int n) { return dim3((n + 255) / 256); };

    // ---- weight conversions f32 -> bf16 ----
    cvt_f32_to_bf16<<<g1(NE * 4 * NH), blk, 0, stream>>>(W_enc,  Wencb,  NE * 4 * NH);
    cvt_f32_to_bf16<<<g1(NH * 4 * NH), blk, 0, stream>>>(U_enc,  Uencb,  NH * 4 * NH);
    cvt_f32_to_bf16<<<g1((NE + NA) * 4 * NH), blk, 0, stream>>>(W_dec, Wdecb, (NE + NA) * 4 * NH);
    cvt_f32_to_bf16<<<g1(NH * 4 * NH), blk, 0, stream>>>(U_dec,  Udecb,  NH * 4 * NH);
    cvt_f32_to_bf16<<<g1(NH * NA),     blk, 0, stream>>>(W_mem,  Wmemb,  NH * NA);
    cvt_f32_to_bf16<<<g1(2 * NH * NA), blk, 0, stream>>>(W_attn, Wattnb, 2 * NH * NA);
    cvt_f32_to_bf16<<<g1(NA * NV),     blk, 0, stream>>>(W_out,  Woutb,  NA * NV);

    fill_zero_u32<<<g1(NB * NH / 2), blk, 0, stream>>>((unsigned int*)zerobf, NB * NH / 2);
    fill_zero_u32<<<g1(NB * NH),     blk, 0, stream>>>((unsigned int*)c_enc,  NB * NH);

    // ---- encoder: embed + precompute x@W_enc for all timesteps ----
    enc_embed<<<g1(NB * NS * NE), blk, 0, stream>>>(enc_ids, enc_emb, xe, NB * NS * NE);
    {
        dim3 grid(4 * NH / 64, NB * NS / 16);
        gemm_bf16_wmma<<<grid, dim3(128), 0, stream>>>(
            xe, NE, Wencb, 4 * NH, gatesx, 4 * NH, nullptr, 0, nullptr,
            NB * NS, 4 * NH, NE);
    }

    // ---- encoder recurrence (128 sequential steps) ----
    for (int t = 0; t < NS; ++t) {
        const bf16_t* Ah = (t == 0) ? zerobf : (encoutb + (size_t)(t - 1) * NH);
        int lda = (t == 0) ? NH : NS * NH;
        dim3 grid(4 * NH / 64, NB / 16);
        gemm_bf16_wmma<<<grid, dim3(128), 0, stream>>>(
            Ah, lda, Uencb, 4 * NH, z_enc, 4 * NH,
            gatesx + (size_t)t * 4 * NH, NS * 4 * NH, nullptr,
            NB, 4 * NH, NH);
        lstm_cell<<<g1(NB * NH), blk, 0, stream>>>(
            z_enc, b_enc, c_enc, c_enc,
            encout + (size_t)t * NH, NS * NH,
            encoutb + (size_t)t * NH, NS * NH,
            NB, NH);
    }

    // ---- attention keys: enc_out @ W_mem ----
    {
        dim3 grid(NA / 64, NB * NS / 16);
        gemm_bf16_wmma<<<grid, dim3(128), 0, stream>>>(
            encoutb, NH, Wmemb, NA, keys, NA, nullptr, 0, nullptr,
            NB * NS, NA, NH);
    }

    // ---- beam search init ----
    dec_init<<<g1(NM * NH), blk, 0, stream>>>(encout, c_enc, hbf, c_state, attn_st,
                                              cum0, fin0, tok);

    // ---- decoder loop (48 sequential steps) ----
    for (int t = 0; t < NT; ++t) {
        float* cumA = (t & 1) ? cum1 : cum0;
        float* cumB = (t & 1) ? cum0 : cum1;
        int*   finA = (t & 1) ? fin1 : fin0;
        int*   finB = (t & 1) ? fin0 : fin1;

        dec_embed_cat<<<g1(NM * (NE + NA)), blk, 0, stream>>>(tok, dec_emb, attn_st, inp);
        {
            dim3 grid(4 * NH / 64, NM / 16);
            gemm_bf16_wmma<<<grid, dim3(128), 0, stream>>>(
                inp, NE + NA, Wdecb, 4 * NH, z_dec, 4 * NH, nullptr, 0, nullptr,
                NM, 4 * NH, NE + NA);
            gemm_bf16_wmma<<<grid, dim3(128), 0, stream>>>(
                hbf, NH, Udecb, 4 * NH, z_dec, 4 * NH, z_dec, 4 * NH, nullptr,
                NM, 4 * NH, NH);
        }
        lstm_cell<<<g1(NM * NH), blk, 0, stream>>>(
            z_dec, b_dec, c_state, c_new, cellout, NH, nullptr, 0, NM, NH);

        attn_scores_softmax<<<dim3(NM), dim3(NS), 0, stream>>>(cellout, keys, align);
        ctx_cat<<<g1(NM * NH), blk, 0, stream>>>(align, encout, cellout, cat2);
        {
            dim3 grid(NA / 64, NM / 16);
            gemm_bf16_wmma<<<grid, dim3(128), 0, stream>>>(
                cat2, 2 * NH, Wattnb, NA, attnnew, NA, nullptr, 0, nullptr,
                NM, NA, 2 * NH);
        }
        cvt_f32_to_bf16<<<g1(NM * NA), blk, 0, stream>>>(attnnew, attnnb, NM * NA);
        {
            dim3 grid(NV / 64, NM / 16);
            gemm_bf16_wmma<<<grid, dim3(128), 0, stream>>>(
                attnnb, NA, Woutb, NV, logits, NV, nullptr, 0, b_out,
                NM, NV, NA);
        }
        log_softmax_rows<<<dim3(NM), dim3(256), 0, stream>>>(logits);
        beam_topk<<<dim3(NB), dim3(256), 0, stream>>>(
            logits, cumA, finA, cumB, finB, tok,
            parents + (size_t)t * NM, toks + (size_t)t * NM);
        beam_gather<<<g1(NM * NH), blk, 0, stream>>>(
            parents + (size_t)t * NM, cellout, c_new, attnnew,
            c_state, attn_st, hbf);
    }

    // ---- backtrack beam ancestry into d_out [B, T, BEAM] int32 ----
    gather_tree<<<dim3(1), dim3(NM), 0, stream>>>(parents, toks, (int*)d_out);
}
